// Experts_83605833383970
// MI455X (gfx1250) — matrix operand order repaired and checked by reading, not compiled, not run on hardware
//
#include <hip/hip_runtime.h>

// ---------------------------------------------------------------------------
// Grouped expert MLP: out = gelu(x @ w1 + b1) @ w2 + b2, per (b,e) slice.
// B=2 E=8 N=1024 D=1024 H=4096, all inputs f32.
//
// Pipeline (all in d_ws):
//   P0: x  f32 -> bf16 row-major                      (33.5 MB)
//   P1: w1 f32 -> bf16 fragment-packed [e][nt][kt][lane][16]   (64 MB)
//   P2: w2 f32 -> bf16 fragment-packed                          (64 MB)
//   G1: h   = gelu(xb @ w1p + b1)  -> bf16 row-major  (128 MB)
//   G2: out = h @ w2p + b2         -> f32
// GEMM hot loop has zero conversion VALU: staging is pure 16B async copies
// (GLOBAL_LOAD_ASYNC_TO_LDS_B128), fragments come from LDS as b128s.
// ---------------------------------------------------------------------------

typedef __attribute__((ext_vector_type(16))) __bf16 v16bf;
typedef __attribute__((ext_vector_type(8)))  float  v8f;
typedef int v4i __attribute__((vector_size(16)));

union Frag { v16bf v; unsigned short s[16]; uint4 q[2]; };

#if __has_builtin(__builtin_amdgcn_global_load_async_to_lds_b128)
#define USE_ASYNC 1
#else
#define USE_ASYNC 0
#endif

__device__ __forceinline__ void stage16(void* lds, const void* g) {
#if USE_ASYNC
    __builtin_amdgcn_global_load_async_to_lds_b128(
        (__attribute__((address_space(1))) v4i*)(unsigned long long)g,
        (__attribute__((address_space(3))) v4i*)lds,
        0, 0);
#else
    *(uint4*)lds = *(const uint4*)g;
#endif
}

__device__ __forceinline__ void wait_async0() {
#if USE_ASYNC
#if __has_builtin(__builtin_amdgcn_s_wait_asynccnt)
    __builtin_amdgcn_s_wait_asynccnt(0);
#else
    asm volatile("s_wait_asynccnt 0" ::: "memory");
#endif
#endif
}

__device__ __forceinline__ unsigned short f2bf(float f) {
    union { float f; unsigned u; } c; c.f = f;
    unsigned u = c.u + 0x7FFFu + ((c.u >> 16) & 1u);   // RNE
    return (unsigned short)(u >> 16);
}
__device__ __forceinline__ unsigned pk2(float lo, float hi) {
    return (unsigned)f2bf(lo) | ((unsigned)f2bf(hi) << 16);
}

__device__ __forceinline__ float gelu_tanh(float x) {
    float t = tanhf(0.7978845608028654f * (x + 0.044715f * x * x * x));
    return 0.5f * x * (1.0f + t);
}

__device__ __forceinline__ void st_elem(float* p, float v) { *p = v; }
__device__ __forceinline__ void st_elem(unsigned short* p, float v) { *p = f2bf(v); }

// --------------------------- prep kernels ----------------------------------

// f32 -> bf16 row-major, 8 elems/thread. n must be a multiple of 2048.
__global__ __launch_bounds__(256)
void cvt_f32_bf16(const float* __restrict__ s, unsigned short* __restrict__ d) {
    const long i = ((long)blockIdx.x * 256 + threadIdx.x) * 8;
    const float4 a = *(const float4*)(s + i);
    const float4 b = *(const float4*)(s + i + 4);
    uint4 u;
    u.x = pk2(a.x, a.y); u.y = pk2(a.z, a.w);
    u.z = pk2(b.x, b.y); u.w = pk2(b.z, b.w);
    *(uint4*)(d + i) = u;
}

// W [e][K][N] f32 -> Wp fragment-packed bf16: [e][nt][kt][lane(32)][16].
// WMMA 16-bit B 32x16 layout: lane<16 -> K={0..7,16..23}, lane>=16 -> K={8..15,24..31}; N=lane%16.
__global__ __launch_bounds__(256)
void pack_w(const float* __restrict__ W, unsigned short* __restrict__ Wp,
            int K, int N) {
    const long gid  = (long)blockIdx.x * 256 + threadIdx.x;
    const int  lane = (int)(gid & 31);
    const long frag = gid >> 5;
    const int  KT = K >> 5, NT = N >> 4;
    const int  kt = (int)(frag % KT);
    const long t1 = frag / KT;
    const int  nt = (int)(t1 % NT);
    const int  e  = (int)(t1 / NT);
    const float* src = W + (long)e * K * N
                         + (long)(kt * 32 + ((lane & 16) >> 1)) * N
                         + nt * 16 + (lane & 15);
    Frag f;
#pragma unroll
    for (int v = 0; v < 16; ++v) f.s[v] = f2bf(src[(long)(v + (v & 8)) * N]);
    *(v16bf*)(Wp + gid * 16) = f.v;
}

// --------------------------- GEMM kernel -----------------------------------
// C[z] = act(A[z] @ W[e] + bias[e]);  A bf16 row-major, W fragment-packed.
// 256 threads (8 wave32), 128x128 tile, K-step 32, double-buffered LDS.
template <typename OT, bool GELU>
__global__ __launch_bounds__(256)
void gemm_bf16_wmma(const unsigned short* __restrict__ A,
                    const unsigned short* __restrict__ Wp,
                    const float* __restrict__ bias,
                    OT* __restrict__ C,
                    int M, int K, int N, int E) {
    __shared__ v16bf As[2][8][32];   // 8 KB / buffer
    __shared__ v16bf Bs[2][8][32];

    const int tid  = threadIdx.x;
    const int lane = tid & 31;
    const int wave = tid >> 5;
    const int wr   = wave >> 2;          // 0..1
    const int wc   = wave & 3;           // 0..3
    const int m0   = blockIdx.y * 128;
    const int nt0  = blockIdx.x * 8;     // n0/16
    const int n0   = blockIdx.x * 128;
    const int KT   = K >> 5, NT = N >> 4;

    const int z = blockIdx.z;
    const int e = z % E;
    const unsigned short* Ab  = A  + (long)z * M * K;
    const unsigned short* Wpb = Wp + (long)e * NT * KT * 512;
    const float*          bb  = bias + (long)e * N;
    OT*                   Cb  = C  + (long)z * M * N;

    // Per-thread staging coords: thread stages A-tile t and B-tile t, lane l.
    // 16-bit A 16x32 layout: lane<16 -> M=lane, K={0..7,16..23}; lane>=16 +8.
    const int t = tid >> 5;
    const int arow = m0 + t * 16 + (lane & 15);
    const unsigned short* aptr = Ab + (long)arow * K + ((lane & 16) >> 1);

    v8f acc[4][2] = {};

    {   // stage k=0 into buffer 0
        stage16((char*)&As[0][t][lane],      aptr);
        stage16((char*)&As[0][t][lane] + 16, aptr + 16);
        const unsigned short* bp = Wpb + (((long)(nt0 + t) * KT) * 32 + lane) * 16;
        stage16((char*)&Bs[0][t][lane],      bp);
        stage16((char*)&Bs[0][t][lane] + 16, bp + 8);
    }

    int buf = 0;
    for (int k0 = 0; k0 < K; k0 += 32) {
        wait_async0();
        __syncthreads();

        const int kn = k0 + 32;
        if (kn < K) {   // prefetch next slab into other buffer (overlaps WMMA)
            const unsigned short* ap = aptr + kn;
            stage16((char*)&As[buf ^ 1][t][lane],      ap);
            stage16((char*)&As[buf ^ 1][t][lane] + 16, ap + 16);
            const unsigned short* bp =
                Wpb + (((long)(nt0 + t) * KT + (kn >> 5)) * 32 + lane) * 16;
            stage16((char*)&Bs[buf ^ 1][t][lane],      bp);
            stage16((char*)&Bs[buf ^ 1][t][lane] + 16, bp + 8);
        }

        const v16bf b0 = Bs[buf][wc * 2 + 0][lane];
        const v16bf b1 = Bs[buf][wc * 2 + 1][lane];
#pragma unroll
        for (int i = 0; i < 4; ++i) {
            const v16bf a = As[buf][wr * 4 + i][lane];
            acc[i][0] = __builtin_amdgcn_wmma_f32_16x16x32_bf16(
                false, a, false, b0, (short)0, acc[i][0], false, false);
            acc[i][1] = __builtin_amdgcn_wmma_f32_16x16x32_bf16(
                false, a, false, b1, (short)0, acc[i][1], false, false);
        }
        buf ^= 1;
    }

    // Epilogue: C 16x16 f32 layout -> VGPR r: M = r + 8*(lane>=16), N = lane%16.
#pragma unroll
    for (int i = 0; i < 4; ++i) {
#pragma unroll
        for (int j = 0; j < 2; ++j) {
            const int col   = n0 + wc * 32 + j * 16 + (lane & 15);
            const int rbase = m0 + wr * 64 + i * 16 + ((lane & 16) >> 1);
            const float bv  = bb[col];
#pragma unroll
            for (int r = 0; r < 8; ++r) {
                float x = acc[i][j][r] + bv;
                if (GELU) x = gelu_tanh(x);
                st_elem(Cb + (long)(rbase + r) * N + col, x);
            }
        }
    }
}

// ---------------------------------------------------------------------------
extern "C" void kernel_launch(void* const* d_in, const int* in_sizes, int n_in,
                              void* d_out, int out_size, void* d_ws, size_t ws_size,
                              hipStream_t stream) {
    constexpr int B = 2, E = 8, N = 1024, D = 1024, H = 4096;

    const float* x  = (const float*)d_in[0];
    const float* w1 = (const float*)d_in[1];
    const float* b1 = (const float*)d_in[2];
    const float* w2 = (const float*)d_in[3];
    const float* b2 = (const float*)d_in[4];
    float* out = (float*)d_out;

    // workspace layout (bytes)
    char* ws = (char*)d_ws;
    unsigned short* xb  = (unsigned short*)(ws);                     //  33.5 MB
    unsigned short* w1p = (unsigned short*)(ws + (size_t)33554432);  //  64 MB
    unsigned short* w2p = (unsigned short*)(ws + (size_t)100663296); //  64 MB
    unsigned short* h   = (unsigned short*)(ws + (size_t)167772160); // 128 MB

    const dim3 blk(256);

    // P0: x -> bf16 (16.7M elems, 8/thread)
    cvt_f32_bf16<<<8192, blk, 0, stream>>>(x, xb);
    // P1/P2: weights -> fragment-packed bf16 (E*NT*KT*32 threads each = 2^21)
    pack_w<<<8192, blk, 0, stream>>>(w1, w1p, D, H);
    pack_w<<<8192, blk, 0, stream>>>(w2, w2p, H, D);

    // G1: h = gelu(xb @ w1p + b1)   M=1024 K=1024 Ncols=4096
    gemm_bf16_wmma<unsigned short, true>
        <<<dim3(H / 128, N / 128, B * E), blk, 0, stream>>>(
            xb, w1p, b1, h, N, D, H, E);

    // G2: out = h @ w2p + b2        M=1024 K=4096 Ncols=1024
    gemm_bf16_wmma<float, false>
        <<<dim3(D / 128, N / 128, B * E), blk, 0, stream>>>(
            h, w2p, b2, out, N, H, D, E);
}
